// PrunedAttention_19095424598882
// MI455X (gfx1250) — matrix operand order repaired and checked by reading, not compiled, not run on hardware
//
#include <hip/hip_runtime.h>
#include <hip/hip_bf16.h>

// ---- problem dims (fixed by reference) ----
#define B_     2
#define H_     32
#define HKV_   8
#define T_     1024
#define S_     2048
#define DH_    128
#define GROUPS (H_ / HKV_)      // 4
#define DMODEL (H_ * DH_)       // 4096
#define THRESH 0.000488f

typedef __bf16 bf16_t;
typedef __attribute__((ext_vector_type(16))) __bf16 v16bf;
typedef __attribute__((ext_vector_type(8)))  __bf16 v8bf;
typedef __attribute__((ext_vector_type(4)))  __bf16 v4bf;
typedef __attribute__((ext_vector_type(8)))  float  v8f;

#define KC  32    // K per WMMA step / LDS tile depth
#define LDT 40    // padded LDS k-stride in bf16 elems (80B = 5*16B -> aligned b128 rows)

// ------------------------------------------------------------------
// Kernel 1: sums[b][s] = sum_{h,t} attn[b,h,t,s]  (keep = sums > THRESH*H*T)
// ------------------------------------------------------------------
__global__ __launch_bounds__(256)
void reduce_keep(const float* __restrict__ attn, float* __restrict__ sums) {
    const int s  = blockIdx.x * 256 + threadIdx.x;   // 0..S-1
    const int bh = blockIdx.y;                       // b*H + h
    const int t0 = blockIdx.z * 64;
    const float* p = attn + ((size_t)bh * T_ + t0) * S_ + s;
    float acc = 0.f;
    #pragma unroll 8
    for (int t = 0; t < 64; ++t) acc += p[(size_t)t * S_];
    atomicAdd(sums + (bh / H_) * S_ + s, acc);
}

// ------------------------------------------------------------------
// Kernel 2: fused prune-write + (pruned @ v_full) per (b,h).
//   A = pruned[b,h]  [T x S]  (generated on the fly: attn * keep)
//   B = v[b,h/4]     [S x Dh]
//   C = ctx          [b][t][h*Dh + d]   (f32, to workspace)
// Block: 256 thr = 8 waves, tile 128(M) x 128(N), Kc=32.
// Wave w: 32x64 sub-block at (wm = (w>>1)*32, wn = (w&1)*64) -> 2x4 C tiles.
// ------------------------------------------------------------------
__global__ __launch_bounds__(256)
void gemm1_attnv(const float* __restrict__ attn, const float* __restrict__ v,
                 const float* __restrict__ sums, float* __restrict__ pruned,
                 float* __restrict__ ctx) {
    __shared__ __attribute__((aligned(16))) bf16_t As[128 * LDT];
    __shared__ __attribute__((aligned(16))) bf16_t Bs[128 * LDT];

    const int bh  = blockIdx.y;            // b*H + h
    const int b   = bh / H_;
    const int h   = bh % H_;
    const int t0  = blockIdx.x * 128;
    const int hkv = h / GROUPS;

    const int tid  = threadIdx.x;
    const int lane = tid & 31;
    const int wave = tid >> 5;
    const int hi   = (lane >> 4) & 1;
    const int ln   = lane & 15;
    const int wm   = (wave >> 1) * 32;
    const int wn   = (wave & 1) * 64;

    const float thresh_sum = THRESH * (float)(H_ * T_);

    const size_t attn_base = ((size_t)bh * T_ + t0) * S_;
    const size_t v_base    = (size_t)(b * HKV_ + hkv) * S_ * DH_;
    const float* sumrow    = sums + b * S_;

    v8f acc[2][4];
    #pragma unroll
    for (int i = 0; i < 2; ++i)
        #pragma unroll
        for (int j = 0; j < 4; ++j) acc[i][j] = (v8f)0.f;

    for (int k0 = 0; k0 < S_; k0 += KC) {
        // ---- A tile: 128 t-rows x 32 s, fused keep-mask + pruned store ----
        #pragma unroll
        for (int it = 0; it < 4; ++it) {
            const int idx = tid + it * 256;     // 0..1023
            const int row = idx >> 3;           // t-row 0..127
            const int cg  = idx & 7;            // 4-float col group
            const size_t g = attn_base + (size_t)row * S_ + k0 + cg * 4;
            float4 a  = *(const float4*)(attn + g);
            float4 kp = *(const float4*)(sumrow + k0 + cg * 4);
            a.x = (kp.x > thresh_sum) ? a.x : 0.f;
            a.y = (kp.y > thresh_sum) ? a.y : 0.f;
            a.z = (kp.z > thresh_sum) ? a.z : 0.f;
            a.w = (kp.w > thresh_sum) ? a.w : 0.f;
            *(float4*)(pruned + g) = a;         // pruned_weights output (write-once)
            v4bf ab = { (bf16_t)a.x, (bf16_t)a.y, (bf16_t)a.z, (bf16_t)a.w };
            *(v4bf*)(&As[row * LDT + cg * 4]) = ab;
        }
        // ---- B tile transposed: Bs[d][k] = v[s=k0+k][d] ----
        #pragma unroll
        for (int it = 0; it < 4; ++it) {
            const int idx = tid + it * 256;
            const int kk  = idx >> 5;           // k 0..31
            const int cg  = idx & 31;           // 4-d group
            float4 b4 = *(const float4*)(v + v_base + (size_t)(k0 + kk) * DH_ + cg * 4);
            Bs[(cg * 4 + 0) * LDT + kk] = (bf16_t)b4.x;
            Bs[(cg * 4 + 1) * LDT + kk] = (bf16_t)b4.y;
            Bs[(cg * 4 + 2) * LDT + kk] = (bf16_t)b4.z;
            Bs[(cg * 4 + 3) * LDT + kk] = (bf16_t)b4.w;
        }
        __syncthreads();

        // ---- fragments (per ISA 16-bit A 16x32 / B 32x16 layouts) ----
        v16bf afrag[2], bfrag[4];
        #pragma unroll
        for (int mt = 0; mt < 2; ++mt) {
            const bf16_t* base = &As[(wm + mt * 16 + ln) * LDT + 8 * hi];
            v8bf lo = *(const v8bf*)(base);        // K = 8hi .. +7
            v8bf up = *(const v8bf*)(base + 16);   // K = 16+8hi .. +7
            afrag[mt] = __builtin_shufflevector(lo, up, 0,1,2,3,4,5,6,7,8,9,10,11,12,13,14,15);
        }
        #pragma unroll
        for (int nt = 0; nt < 4; ++nt) {
            const bf16_t* base = &Bs[(wn + nt * 16 + ln) * LDT + 8 * hi];
            v8bf lo = *(const v8bf*)(base);        // K = 8hi .. +7
            v8bf up = *(const v8bf*)(base + 16);   // K = 16+8hi .. +7
            bfrag[nt] = __builtin_shufflevector(lo, up, 0,1,2,3,4,5,6,7,8,9,10,11,12,13,14,15);
        }
        #pragma unroll
        for (int mt = 0; mt < 2; ++mt)
            #pragma unroll
            for (int nt = 0; nt < 4; ++nt)
                acc[mt][nt] = __builtin_amdgcn_wmma_f32_16x16x32_bf16(
                    false, afrag[mt], false, bfrag[nt], (short)0, acc[mt][nt], false, false);
        __syncthreads();
    }

    // ---- store ctx: [b][t][h*Dh + d] ----
    #pragma unroll
    for (int mt = 0; mt < 2; ++mt)
        #pragma unroll
        for (int nt = 0; nt < 4; ++nt) {
            const int trow = t0 + wm + mt * 16 + 8 * hi;
            const int d    = wn + nt * 16 + ln;
            #pragma unroll
            for (int r = 0; r < 8; ++r)
                ctx[(((size_t)b * T_ + (trow + r)) * H_ + h) * DH_ + d] = acc[mt][nt][r];
        }
}

// ------------------------------------------------------------------
// Kernel 3: out[m][n] = sum_k ctx[m][k] * w_o[n][k]   (M=B*T, N=K=4096)
// Same tiling; w_o rows are naturally [n][k] -> no transpose on B fill.
// ------------------------------------------------------------------
__global__ __launch_bounds__(256)
void gemm2_oproj(const float* __restrict__ ctx, const float* __restrict__ w_o,
                 float* __restrict__ out) {
    __shared__ __attribute__((aligned(16))) bf16_t As[128 * LDT];
    __shared__ __attribute__((aligned(16))) bf16_t Bs[128 * LDT];

    const int m0 = blockIdx.x * 128;       // over B*T = 2048
    const int n0 = blockIdx.y * 128;       // over 4096

    const int tid  = threadIdx.x;
    const int lane = tid & 31;
    const int wave = tid >> 5;
    const int hi   = (lane >> 4) & 1;
    const int ln   = lane & 15;
    const int wm   = (wave >> 1) * 32;
    const int wn   = (wave & 1) * 64;

    v8f acc[2][4];
    #pragma unroll
    for (int i = 0; i < 2; ++i)
        #pragma unroll
        for (int j = 0; j < 4; ++j) acc[i][j] = (v8f)0.f;

    for (int k0 = 0; k0 < DMODEL; k0 += KC) {
        #pragma unroll
        for (int it = 0; it < 4; ++it) {
            const int idx = tid + it * 256;
            const int row = idx >> 3;
            const int cg  = idx & 7;
            float4 a = *(const float4*)(ctx + (size_t)(m0 + row) * DMODEL + k0 + cg * 4);
            v4bf ab = { (bf16_t)a.x, (bf16_t)a.y, (bf16_t)a.z, (bf16_t)a.w };
            *(v4bf*)(&As[row * LDT + cg * 4]) = ab;
            float4 w = *(const float4*)(w_o + (size_t)(n0 + row) * DMODEL + k0 + cg * 4);
            v4bf wb = { (bf16_t)w.x, (bf16_t)w.y, (bf16_t)w.z, (bf16_t)w.w };
            *(v4bf*)(&Bs[row * LDT + cg * 4]) = wb;
        }
        __syncthreads();

        v16bf afrag[2], bfrag[4];
        #pragma unroll
        for (int mt = 0; mt < 2; ++mt) {
            const bf16_t* base = &As[(wm + mt * 16 + ln) * LDT + 8 * hi];
            v8bf lo = *(const v8bf*)(base);
            v8bf up = *(const v8bf*)(base + 16);
            afrag[mt] = __builtin_shufflevector(lo, up, 0,1,2,3,4,5,6,7,8,9,10,11,12,13,14,15);
        }
        #pragma unroll
        for (int nt = 0; nt < 4; ++nt) {
            const bf16_t* base = &Bs[(wn + nt * 16 + ln) * LDT + 8 * hi];
            v8bf lo = *(const v8bf*)(base);
            v8bf up = *(const v8bf*)(base + 16);
            bfrag[nt] = __builtin_shufflevector(lo, up, 0,1,2,3,4,5,6,7,8,9,10,11,12,13,14,15);
        }
        #pragma unroll
        for (int mt = 0; mt < 2; ++mt)
            #pragma unroll
            for (int nt = 0; nt < 4; ++nt)
                acc[mt][nt] = __builtin_amdgcn_wmma_f32_16x16x32_bf16(
                    false, afrag[mt], false, bfrag[nt], (short)0, acc[mt][nt], false, false);
        __syncthreads();
    }

    #pragma unroll
    for (int mt = 0; mt < 2; ++mt)
        #pragma unroll
        for (int nt = 0; nt < 4; ++nt) {
            const int m = m0 + wm + mt * 16 + 8 * hi;
            const int n = n0 + wn + nt * 16 + ln;
            #pragma unroll
            for (int r = 0; r < 8; ++r)
                out[(size_t)(m + r) * DMODEL + n] = acc[mt][nt][r];
        }
}

// ------------------------------------------------------------------
extern "C" void kernel_launch(void* const* d_in, const int* in_sizes, int n_in,
                              void* d_out, int out_size, void* d_ws, size_t ws_size,
                              hipStream_t stream) {
    (void)in_sizes; (void)n_in; (void)out_size; (void)ws_size;
    const float* attn = (const float*)d_in[0];   // [B,H,T,S]
    const float* v    = (const float*)d_in[1];   // [B,HKV,S,Dh]
    const float* w_o  = (const float*)d_in[2];   // [DMODEL,DMODEL]

    float* out    = (float*)d_out;                           // [B,T,DMODEL]
    float* pruned = out + (size_t)B_ * T_ * DMODEL;          // [B,H,T,S]

    float* sums = (float*)d_ws;                              // [B,S]
    float* ctx  = sums + B_ * S_;                            // [B,T,H*Dh] f32 (16KB offset, aligned)

    hipMemsetAsync(sums, 0, (size_t)B_ * S_ * sizeof(float), stream);

    dim3 rg(S_ / 256, B_ * H_, T_ / 64);
    reduce_keep<<<rg, 256, 0, stream>>>(attn, sums);

    dim3 g1(T_ / 128, B_ * H_);
    gemm1_attnv<<<g1, 256, 0, stream>>>(attn, v, sums, pruned, ctx);

    dim3 g2((B_ * T_) / 128, DMODEL / 128);
    gemm2_oproj<<<g2, 256, 0, stream>>>(ctx, w_o, out);
}
